// MDPCLassifier_29068338659821
// MI455X (gfx1250) — matrix operand order repaired and checked by reading, not compiled, not run on hardware
//
#include <hip/hip_runtime.h>
#include <math.h>

// Problem constants (match reference)
#define BB 16384
#define TT 30
#define EE 20
#define HH 256
#define GG 1024   // 4*H
#define AA 64
#define CCOUT 512

typedef __attribute__((ext_vector_type(16))) __bf16 bf16x16;
typedef __attribute__((ext_vector_type(8)))  float  f32x8;
typedef __attribute__((ext_vector_type(4)))  unsigned int u32x4;

union ABOp {
    bf16x16 v;
    u32x4 q[2];
    unsigned short us[16];
};

__device__ __forceinline__ unsigned short f2bf(float f) {
    union { float f; unsigned int u; } a; a.f = f;
    unsigned int u = a.u;
    u += 0x7fffu + ((u >> 16) & 1u);   // round-to-nearest-even
    return (unsigned short)(u >> 16);
}
__device__ __forceinline__ float bf2f(unsigned short s) {
    union { unsigned int u; float f; } a; a.u = ((unsigned int)s) << 16;
    return a.f;
}
__device__ __forceinline__ float sigm(float x) { return 1.0f / (1.0f + __expf(-x)); }
__device__ __forceinline__ float tanh_fast(float x) {
    x = fminf(15.0f, fmaxf(-15.0f, x));
    float e = __expf(2.0f * x);
    return (e - 1.0f) / (e + 1.0f);
}
__device__ __forceinline__ f32x8 splat8(float v) {
    f32x8 r;
#pragma unroll
    for (int i = 0; i < 8; i++) r[i] = v;
    return r;
}
#define WMMA_BF16(A, B, C) \
    __builtin_amdgcn_wmma_f32_16x16x32_bf16(false, (A), false, (B), (short)0, (C), false, false)

// ---------------------------------------------------------------------------
// Workspace layout (bytes). Matrix weights are bf16, pre-swizzled into the
// exact WMMA B-operand lane layout: tile (nt,kt): lane l supplies
// N = nt*16+(l&15), K = kt*32 + 8*(l>>4) + j + 8*(j>>3) (j=0..15), so a
// B-tile fetch is one fully coalesced 1KB wave read.
// gtab[v][n] = emb[v] @ W_ih[n].T + b_ih[n] + b_hh[n]  (token -> gate bias;
// only V=100 tokens exist, so the whole input GEMM collapses to a 400KB
// L2-resident gather that also initializes the accumulators).
// ---------------------------------------------------------------------------
#define OFF_WHHT 0u          // 64nt x 8kt x 32lane x 16 bf16 = 524288 B
#define OFF_GTAB 524288u     // 100 x 1024 f32                = 409600 B
#define OFF_FC1T 933888u     // 4nt x 8kt x 512 bf16          = 32768 B
#define OFF_AWT  966656u     // 4nt x 2kt x 512 bf16          = 8192 B
#define OFF_FC2T 974848u     // 4nt x 4kt x 512 bf16          = 16384 B
#define OFF_FC3T 991232u     // 4nt x 4kt x 512 bf16          = 16384 B
#define OFF_FC4T 1007616u    // 32nt x 2kt x 512 bf16         = 65536 B
#define OFF_ENC  1073152u    // 16384 x 256 bf16              = 8388608 B

// ---------------------------------------------------------------------------
// Prep: weight conversion/swizzle + token gate-bias table
// ---------------------------------------------------------------------------
__global__ __launch_bounds__(256) void prep_kernel(
    const float* __restrict__ emb, const float* __restrict__ Wih,
    const float* __restrict__ Whh, const float* __restrict__ bih,
    const float* __restrict__ bhh, const float* __restrict__ fc1W,
    const float* __restrict__ aW, const float* __restrict__ fc2W,
    const float* __restrict__ fc3W, const float* __restrict__ fc4W,
    char* __restrict__ ws)
{
    unsigned short* whhT = (unsigned short*)(ws + OFF_WHHT);
    float*          gtab = (float*)(ws + OFF_GTAB);
    unsigned short* fc1T = (unsigned short*)(ws + OFF_FC1T);
    unsigned short* awT  = (unsigned short*)(ws + OFF_AWT);
    unsigned short* fc2T = (unsigned short*)(ws + OFF_FC2T);
    unsigned short* fc3T = (unsigned short*)(ws + OFF_FC3T);
    unsigned short* fc4T = (unsigned short*)(ws + OFF_FC4T);

    const int g = blockIdx.x * 256 + threadIdx.x;
    const int stride = gridDim.x * 256;

    // W_hh: 64nt x 8kt B-operand tiles
    for (int i = g; i < 64 * 8 * 32 * 16; i += stride) {
        int j = i & 15, lane = (i >> 4) & 31, kt = (i >> 9) & 7, nt = i >> 12;
        int n = nt * 16 + (lane & 15);
        int k = kt * 32 + 8 * (lane >> 4) + j + 8 * (j >> 3);
        whhT[i] = f2bf(Whh[n * HH + k]);
    }
    // gtab[v][n] = emb[v,:] . Wih[n,:] + b_ih[n] + b_hh[n]
    for (int i = g; i < 100 * GG; i += stride) {
        int v = i >> 10, n = i & (GG - 1);
        float s = bih[n] + bhh[n];
#pragma unroll
        for (int k = 0; k < EE; k++) s += emb[v * EE + k] * Wih[n * EE + k];
        gtab[i] = s;
    }
    // fc1: 4nt x 8kt, K=256
    for (int i = g; i < 4 * 8 * 32 * 16; i += stride) {
        int j = i & 15, lane = (i >> 4) & 31, kt = (i >> 9) & 7, nt = i >> 12;
        fc1T[i] = f2bf(fc1W[(nt * 16 + (lane & 15)) * HH + kt * 32 + 8 * (lane >> 4) + j + 8 * (j >> 3)]);
    }
    // aW: 4nt x 2kt, K=64
    for (int i = g; i < 4 * 2 * 32 * 16; i += stride) {
        int j = i & 15, lane = (i >> 4) & 31, kt = (i >> 9) & 1, nt = i >> 10;
        awT[i] = f2bf(aW[(nt * 16 + (lane & 15)) * AA + kt * 32 + 8 * (lane >> 4) + j + 8 * (j >> 3)]);
    }
    // fc2, fc3: 4nt x 4kt, K=128
    for (int i = g; i < 4 * 4 * 32 * 16; i += stride) {
        int j = i & 15, lane = (i >> 4) & 31, kt = (i >> 9) & 3, nt = i >> 11;
        int src = (nt * 16 + (lane & 15)) * 128 + kt * 32 + 8 * (lane >> 4) + j + 8 * (j >> 3);
        fc2T[i] = f2bf(fc2W[src]);
        fc3T[i] = f2bf(fc3W[src]);
    }
    // fc4: 32nt x 2kt, K=64
    for (int i = g; i < 32 * 2 * 32 * 16; i += stride) {
        int j = i & 15, lane = (i >> 4) & 31, kt = (i >> 9) & 1, nt = i >> 10;
        fc4T[i] = f2bf(fc4W[(nt * 16 + (lane & 15)) * AA + kt * 32 + 8 * (lane >> 4) + j + 8 * (j >> 3)]);
    }
}

// ---------------------------------------------------------------------------
// Persistent LSTM: each block owns 32 batch rows for all T=30 steps; h/c/e
// live in LDS. Wave w owns hidden-column tiles cc in {w, w+8}; gate tiles
// nt = 16*g + cc land in the same wave, so the cell update never crosses
// waves. One barrier per step (h double-buffer swap). The cci loop is kept
// rolled so only one column-half's accumulators/B-tiles are live at a time
// (keeps the wave under 256 VGPRs -> no s_set_vgpr_msb churn).
// ---------------------------------------------------------------------------
__global__ __launch_bounds__(256) void lstm_kernel(
    const int* __restrict__ obs, const char* __restrict__ ws)
{
    const unsigned short* whhT = (const unsigned short*)(ws + OFF_WHHT);
    const float*          gtab = (const float*)(ws + OFF_GTAB);
    unsigned short*       encg = (unsigned short*)((char*)ws + OFF_ENC);

    __shared__ __align__(16) unsigned short sh_h[2][32][264]; // bf16, padded stride
    __shared__ __align__(16) unsigned short sh_e[32][264];    // bf16 encoded blend
    __shared__ __align__(16) float          sh_c[32][264];    // f32 cell state

    const int tid = threadIdx.x;
    const int w = tid >> 5;
    const int lane = tid & 31;
    const int lanelo = lane & 15;
    const int hi = lane >> 4;
    const int b0 = blockIdx.x * 32;

    for (int i = tid; i < 32 * 264; i += 256) {
        (&sh_h[0][0][0])[i] = 0;
        (&sh_e[0][0])[i] = 0;
        (&sh_c[0][0])[i] = 0.0f;
    }
    __syncthreads();

    for (int t = 0; t < TT; t++) {
        const int cur = t & 1, nxt = cur ^ 1;

        // tokens for this step (per m-tile / per-element row); also the mask
        int tokr[2][8];
#pragma unroll
        for (int m = 0; m < 2; m++)
#pragma unroll
            for (int i = 0; i < 8; i++) {
                int row = m * 16 + i + 8 * hi;
                tokr[m][i] = obs[(b0 + row) * TT + t];
            }

#pragma clang loop unroll(disable)
        for (int cci = 0; cci < 2; cci++) {
            const int cc = w + 8 * cci;

            // acc init = token gate-bias gather (x@W_ih.T + b_ih + b_hh folded)
            f32x8 acc[4][2];
#pragma unroll
            for (int m = 0; m < 2; m++)
#pragma unroll
                for (int i = 0; i < 8; i++) {
                    const float* gb = gtab + tokr[m][i] * GG + cc * 16 + lanelo;
#pragma unroll
                    for (int gt = 0; gt < 4; gt++)
                        acc[gt][m][i] = gb[gt * 256];
                }

            // ---- recurrent GEMM: h @ W_hh.T (K=256 -> 8 K-tiles) ----
            const unsigned short* wbase = whhT + (cc * 8 * 32 + lane) * 16;
#pragma unroll
            for (int kt = 0; kt < 8; kt++) {
                ABOp ah[2];
#pragma unroll
                for (int m = 0; m < 2; m++) {
                    int row = m * 16 + lanelo, k0 = kt * 32 + 8 * hi;
                    ah[m].q[0] = *(const u32x4*)&sh_h[cur][row][k0];
                    ah[m].q[1] = *(const u32x4*)&sh_h[cur][row][k0 + 16];
                }
                ABOp bt[4];
#pragma unroll
                for (int gt = 0; gt < 4; gt++) {
                    // gate gt, K-tile kt: constant offset from per-cci base
                    const unsigned short* p = wbase + (gt * 16 * 8 + kt) * 32 * 16;
                    bt[gt].q[0] = *(const u32x4*)p;
                    bt[gt].q[1] = *(const u32x4*)(p + 8);
                }
#pragma unroll
                for (int gt = 0; gt < 4; gt++)
#pragma unroll
                    for (int m = 0; m < 2; m++)
                        acc[gt][m] = WMMA_BF16(ah[m].v, bt[gt].v, acc[gt][m]);
            }

            // ---- LSTM cell (i/f/g/o all owned by this wave; c in LDS) ----
#pragma unroll
            for (int m = 0; m < 2; m++) {
#pragma unroll
                for (int i = 0; i < 8; i++) {
                    float iv = sigm(acc[0][m][i]);
                    float fv = sigm(acc[1][m][i]);
                    float gv = tanh_fast(acc[2][m][i]);
                    float ov = sigm(acc[3][m][i]);
                    int row = m * 16 + i + 8 * hi;
                    int col = cc * 16 + lanelo;
                    float c = fv * sh_c[row][col] + iv * gv;
                    sh_c[row][col] = c;
                    float h = ov * tanh_fast(c);
                    float mt = (tokr[m][i] == 0) ? 1.0f : 0.0f;
                    float eo = bf2f(sh_e[row][col]);
                    sh_e[row][col] = f2bf((1.0f - mt) * h + mt * eo);
                    sh_h[nxt][row][col] = f2bf(h);
                }
            }
        }
        __syncthreads();
    }

    // spill encoded state to workspace for the head kernel
    for (int i = tid; i < 32 * HH; i += 256) {
        int r = i >> 8, c = i & 255;
        encg[(b0 + r) * HH + c] = sh_e[r][c];
    }
}

// ---------------------------------------------------------------------------
// Heads: fc1/aW -> common -> {fc2->relu->fc4->argmax, fc3->relu->fc5}
// 128 blocks x 8 waves; each wave independently handles 16 batch rows.
// ---------------------------------------------------------------------------
__global__ __launch_bounds__(256) void head_kernel(
    const float* __restrict__ actions, const char* __restrict__ ws,
    const float* __restrict__ fc1b, const float* __restrict__ ab,
    const float* __restrict__ fc2b, const float* __restrict__ fc3b,
    const float* __restrict__ fc4b, const float* __restrict__ fc5w,
    const float* __restrict__ fc5b, float* __restrict__ out)
{
    const unsigned short* enc  = (const unsigned short*)(ws + OFF_ENC);
    const unsigned short* fc1T = (const unsigned short*)(ws + OFF_FC1T);
    const unsigned short* awT  = (const unsigned short*)(ws + OFF_AWT);
    const unsigned short* fc2T = (const unsigned short*)(ws + OFF_FC2T);
    const unsigned short* fc3T = (const unsigned short*)(ws + OFF_FC3T);
    const unsigned short* fc4T = (const unsigned short*)(ws + OFF_FC4T);

    __shared__ __align__(16) unsigned short sh_common[8][16][136];
    __shared__ __align__(16) unsigned short sh_h2[8][16][72];

    const int tid = threadIdx.x;
    const int w = tid >> 5;
    const int lane = tid & 31;
    const int lanelo = lane & 15;
    const int hi = lane >> 4;
    const int rb = blockIdx.x * 128 + w * 16;

    // ---- obs_emb = enc @ fc1.T + fc1_b  (cols 0..63 of common) ----
    {
        f32x8 acc[4];
#pragma unroll
        for (int nt = 0; nt < 4; nt++) acc[nt] = splat8(fc1b[nt * 16 + lanelo]);
#pragma unroll
        for (int kt = 0; kt < 8; kt++) {
            ABOp a;
            int row = rb + lanelo, k0 = kt * 32 + 8 * hi;
            a.q[0] = *(const u32x4*)&enc[row * HH + k0];
            a.q[1] = *(const u32x4*)&enc[row * HH + k0 + 16];
#pragma unroll
            for (int nt = 0; nt < 4; nt++) {
                ABOp bt;
                const unsigned short* p = fc1T + ((nt * 8 + kt) * 32 + lane) * 16;
                bt.q[0] = *(const u32x4*)p;
                bt.q[1] = *(const u32x4*)(p + 8);
                acc[nt] = WMMA_BF16(a.v, bt.v, acc[nt]);
            }
        }
#pragma unroll
        for (int nt = 0; nt < 4; nt++)
#pragma unroll
            for (int i = 0; i < 8; i++)
                sh_common[w][i + 8 * hi][nt * 16 + lanelo] = f2bf(acc[nt][i]);
    }
    // ---- action_emb = actions @ aW.T + ab  (cols 64..127 of common) ----
    {
        f32x8 acc[4];
#pragma unroll
        for (int nt = 0; nt < 4; nt++) acc[nt] = splat8(ab[nt * 16 + lanelo]);
#pragma unroll
        for (int kt = 0; kt < 2; kt++) {
            ABOp a;
            int row = rb + lanelo, k0 = kt * 32 + 8 * hi;
#pragma unroll
            for (int j = 0; j < 8; j++) {
                a.us[j]     = f2bf(actions[row * AA + k0 + j]);
                a.us[8 + j] = f2bf(actions[row * AA + k0 + 16 + j]);
            }
#pragma unroll
            for (int nt = 0; nt < 4; nt++) {
                ABOp bt;
                const unsigned short* p = awT + ((nt * 2 + kt) * 32 + lane) * 16;
                bt.q[0] = *(const u32x4*)p;
                bt.q[1] = *(const u32x4*)(p + 8);
                acc[nt] = WMMA_BF16(a.v, bt.v, acc[nt]);
            }
        }
#pragma unroll
        for (int nt = 0; nt < 4; nt++)
#pragma unroll
            for (int i = 0; i < 8; i++)
                sh_common[w][i + 8 * hi][64 + nt * 16 + lanelo] = f2bf(acc[nt][i]);
    }
    __syncthreads();

    // ---- hidden2 = relu(common @ fc2.T + fc2_b) ----
    {
        f32x8 acc[4];
#pragma unroll
        for (int nt = 0; nt < 4; nt++) acc[nt] = splat8(fc2b[nt * 16 + lanelo]);
#pragma unroll
        for (int kt = 0; kt < 4; kt++) {
            ABOp a;
            int k0 = kt * 32 + 8 * hi;
            a.q[0] = *(const u32x4*)&sh_common[w][lanelo][k0];
            a.q[1] = *(const u32x4*)&sh_common[w][lanelo][k0 + 16];
#pragma unroll
            for (int nt = 0; nt < 4; nt++) {
                ABOp bt;
                const unsigned short* p = fc2T + ((nt * 4 + kt) * 32 + lane) * 16;
                bt.q[0] = *(const u32x4*)p;
                bt.q[1] = *(const u32x4*)(p + 8);
                acc[nt] = WMMA_BF16(a.v, bt.v, acc[nt]);
            }
        }
#pragma unroll
        for (int nt = 0; nt < 4; nt++)
#pragma unroll
            for (int i = 0; i < 8; i++)
                sh_h2[w][i + 8 * hi][nt * 16 + lanelo] = f2bf(fmaxf(acc[nt][i], 0.0f));
    }
    __syncthreads();

    // ---- cluster_logits = hidden2 @ fc4.T + fc4_b; streamed argmax ----
    {
        float best[8];
        int bidx[8];
#pragma unroll
        for (int i = 0; i < 8; i++) { best[i] = -3.4e38f; bidx[i] = 0; }

        for (int nt = 0; nt < 32; nt++) {
            f32x8 acc = splat8(fc4b[nt * 16 + lanelo]);
#pragma unroll
            for (int kt = 0; kt < 2; kt++) {
                ABOp a;
                int k0 = kt * 32 + 8 * hi;
                a.q[0] = *(const u32x4*)&sh_h2[w][lanelo][k0];
                a.q[1] = *(const u32x4*)&sh_h2[w][lanelo][k0 + 16];
                ABOp bt;
                const unsigned short* p = fc4T + ((nt * 2 + kt) * 32 + lane) * 16;
                bt.q[0] = *(const u32x4*)p;
                bt.q[1] = *(const u32x4*)(p + 8);
                acc = WMMA_BF16(a.v, bt.v, acc);
            }
            int col = nt * 16 + lanelo;
#pragma unroll
            for (int i = 0; i < 8; i++) {
                int row = rb + i + 8 * hi;
                out[row * CCOUT + col] = acc[i];
                if (acc[i] > best[i]) { best[i] = acc[i]; bidx[i] = col; }
            }
        }
        // argmax reduce across the 16 lanes of each half (wave32, width 16)
#pragma unroll
        for (int off = 8; off >= 1; off >>= 1) {
#pragma unroll
            for (int i = 0; i < 8; i++) {
                float ov = __shfl_xor(best[i], off, 16);
                int   oi = __shfl_xor(bidx[i], off, 16);
                if (ov > best[i] || (ov == best[i] && oi < bidx[i])) {
                    best[i] = ov; bidx[i] = oi;
                }
            }
        }
        if (lanelo == 0) {
#pragma unroll
            for (int i = 0; i < 8; i++)
                out[BB * CCOUT + rb + i + 8 * hi] = (float)bidx[i];
        }
    }

    // ---- reward = relu(common @ fc3.T + fc3_b) @ fc5.T + fc5_b ----
    {
        float rsum[8];
#pragma unroll
        for (int i = 0; i < 8; i++) rsum[i] = 0.0f;
#pragma unroll
        for (int nt = 0; nt < 4; nt++) {
            f32x8 acc = splat8(fc3b[nt * 16 + lanelo]);
#pragma unroll
            for (int kt = 0; kt < 4; kt++) {
                ABOp a;
                int k0 = kt * 32 + 8 * hi;
                a.q[0] = *(const u32x4*)&sh_common[w][lanelo][k0];
                a.q[1] = *(const u32x4*)&sh_common[w][lanelo][k0 + 16];
                ABOp bt;
                const unsigned short* p = fc3T + ((nt * 4 + kt) * 32 + lane) * 16;
                bt.q[0] = *(const u32x4*)p;
                bt.q[1] = *(const u32x4*)(p + 8);
                acc = WMMA_BF16(a.v, bt.v, acc);
            }
            float wv = fc5w[nt * 16 + lanelo];
#pragma unroll
            for (int i = 0; i < 8; i++) rsum[i] += fmaxf(acc[i], 0.0f) * wv;
        }
#pragma unroll
        for (int off = 8; off >= 1; off >>= 1)
#pragma unroll
            for (int i = 0; i < 8; i++) rsum[i] += __shfl_xor(rsum[i], off, 16);
        if (lanelo == 0) {
#pragma unroll
            for (int i = 0; i < 8; i++)
                out[BB * CCOUT + BB + rb + i + 8 * hi] = rsum[i] + fc5b[0];
        }
    }
}

extern "C" void kernel_launch(void* const* d_in, const int* in_sizes, int n_in,
                              void* d_out, int out_size, void* d_ws, size_t ws_size,
                              hipStream_t stream) {
    const int*   obs     = (const int*)d_in[0];
    const float* actions = (const float*)d_in[1];
    const float* emb     = (const float*)d_in[2];
    const float* Wih     = (const float*)d_in[3];
    const float* Whh     = (const float*)d_in[4];
    const float* bih     = (const float*)d_in[5];
    const float* bhh     = (const float*)d_in[6];
    const float* aW      = (const float*)d_in[7];
    const float* ab      = (const float*)d_in[8];
    const float* fc1W    = (const float*)d_in[9];
    const float* fc1b    = (const float*)d_in[10];
    const float* fc2W    = (const float*)d_in[11];
    const float* fc2b    = (const float*)d_in[12];
    const float* fc3W    = (const float*)d_in[13];
    const float* fc3b    = (const float*)d_in[14];
    const float* fc4W    = (const float*)d_in[15];
    const float* fc4b    = (const float*)d_in[16];
    const float* fc5W    = (const float*)d_in[17];
    const float* fc5b    = (const float*)d_in[18];
    char* ws = (char*)d_ws;

    prep_kernel<<<1024, 256, 0, stream>>>(emb, Wih, Whh, bih, bhh, fc1W, aW,
                                          fc2W, fc3W, fc4W, ws);
    lstm_kernel<<<BB / 32, 256, 0, stream>>>(obs, ws);
    head_kernel<<<BB / 128, 256, 0, stream>>>(actions, ws, fc1b, ab, fc2b, fc3b,
                                              fc4b, fc5W, fc5b, (float*)d_out);
}
